// MultiScaleVQVAE_4294967296505
// MI455X (gfx1250) — matrix operand order repaired and checked by reading, not compiled, not run on hardware
//
#include <hip/hip_runtime.h>
#include <cstdint>
#include <cstddef>

// ---------------------------------------------------------------------------
// CDNA5 / gfx1250 MultiScaleVQVAE forward.
// All GEMM-shaped work (convs, conv-transpose via lhs-dilation, 1x1
// projections, VQ distance matmul) goes through one implicit-im2col WMMA
// GEMM using V_WMMA_F32_16X16X32_F16 (wave32), 128x128x32 block tiles,
// 8 WMMAs / wave / K-step, ping-pong LDS + register staging so the next
// tile's (unconditional, clamped-address) global gathers overlap the WMMAs.
// ---------------------------------------------------------------------------

typedef __attribute__((ext_vector_type(16))) _Float16 v16h;
typedef __attribute__((ext_vector_type(8)))  _Float16 v8h;
typedef __attribute__((ext_vector_type(8)))  float    v8f;

#define BM 128
#define BN 128
#define BK 32

__device__ __forceinline__ float silu_f(float v) { return v / (1.f + __expf(-v)); }

// ---------------------------------------------------------------------------
// Pack conv weight (OIHW f32) -> N-major f16 matrix Wp[n][k], n padded to Nr,
// k = ci*KH*KW + r*KW + s padded to Kp. flipT packs
// flip(w,(2,3)).transpose(1,0,2,3) for the lhs-dilated conv-transpose.
// ---------------------------------------------------------------------------
__global__ void __launch_bounds__(256)
pack_weight_kernel(const float* __restrict__ w, _Float16* __restrict__ dst,
                   int N, int Cin, int KH, int KW, int Kp, int Nr, int flipT)
{
    const int t = blockIdx.x * blockDim.x + threadIdx.x;
    const int total = Nr * Kp;
    if (t >= total) return;
    const int n = t / Kp, k = t - n * Kp;
    const int KHW = KH * KW, Kre = Cin * KHW;
    float v = 0.f;
    if (n < N && k < Kre) {
        const int ci = k / KHW;
        const int rs = k - ci * KHW;
        const int r  = rs / KW;
        const int s  = rs - r * KW;
        const long src = flipT
            ? ((((long)ci * N + n) * KH + (KH - 1 - r)) * KW + (KW - 1 - s))
            : ((((long)n * Cin + ci) * KH + r) * KW + s);
        v = w[src];
    }
    dst[t] = (_Float16)v;
}

// ---------------------------------------------------------------------------
// Implicit-im2col WMMA GEMM, C[M x N] = im2col(X) * Wp^T.
// Block tile 128x128x32, 256 threads = 8 waves (4 in M x 2 in N), each wave
// computes 32x64 = 8 x V_WMMA_F32_16X16X32_F16 per K-step.
// A tile stored in LDS in row-major, k-contiguous order so every fragment is
// one contiguous 32B v16h. Ping-pong LDS: one barrier per K-step.
// Gathers load unconditionally from a clamped address (single wait, after
// the WMMAs); f32->f16 conversion deferred to the LDS store.
// Epilogue: +bias[n], +residual, SiLU; storeMode 0 = MxN, 1 = NCHW scatter.
// ---------------------------------------------------------------------------
template <int KHW_T, int KW_T, int DIL_T>
__global__ void __launch_bounds__(256)
wmma_conv_gemm(const float* __restrict__ X, int Bn, int Cin, int H, int W,
               int KH, int KW, int stride, int pad, int dil,
               int OH, int OW,
               const _Float16* __restrict__ Wp, int N, int Kp,
               float* __restrict__ Co, int storeMode,
               const float* __restrict__ residual,
               const float* __restrict__ bias, int act)
{
    __shared__ __align__(32) _Float16 As[2][BM * BK];  // m-major, k contiguous
    __shared__ __align__(32) _Float16 Bs[2][BN * BK];  // n-major, k contiguous

    const int HW   = OH * OW;
    const int M    = Bn * HW;
    const int KHWr = KH * KW;
    const int Kre  = Cin * KHWr;

    const int bm = blockIdx.x * BM;
    const int bn = blockIdx.y * BN;
    const int tid  = threadIdx.x;
    const int lane = tid & 31;           // wave32
    const int wv   = tid >> 5;           // 8 waves
    const int wm   = wv >> 1;            // 0..3 : 32-row slice
    const int wn   = wv & 1;             // 0..1 : 64-col slice

    const v8f vzero = {};
    v8f acc[2][4];
    #pragma unroll
    for (int u = 0; u < 2; ++u)
        #pragma unroll
        for (int t = 0; t < 4; ++t) acc[u][t] = vzero;

    // ---- staging coordinates: each thread stages 16 k's of one A row and
    //      16 k's of one B row per K-step.
    const int ra   = tid >> 1;            // 0..127
    const int kh16 = (tid & 1) << 4;      // 0 or 16
    const int mg   = bm + ra;
    const bool mOK = mg < M;
    int bi = 0, oh = 0, ow = 0;
    if (mOK) { bi = mg / HW; int pix = mg - bi * HW; oh = pix / OW; ow = pix - oh * OW; }
    const int ohs = oh * stride - pad;
    const int ows = ow * stride - pad;
    const float* __restrict__ Xb = X + (long)bi * Cin * H * W;
    const long brow = (long)(bn + ra) * Kp;       // bn+ra < Nr by construction

    // store positions: contiguous k order (row offset = k), matching Bs
    // k-group g (8 k's) -> row offset g<<3
    const int g0 = kh16 >> 3;                       // 0 or 2
    const int p0 = kh16;
    const int p1 = kh16 + 8;

    float aF[16];        // staged A values, f32 until the LDS store
    v16h  bReg;

    auto gatherA = [&](int k0) {
        #pragma unroll
        for (int e = 0; e < 16; ++e) {
            const int k = k0 + kh16 + e;
            int ci, r, sc;
            if constexpr (KHW_T == 1) {
                ci = k; r = 0; sc = 0;
            } else if constexpr (KHW_T > 1) {
                ci = k / KHW_T; const int rs = k - ci * KHW_T;
                r = rs / KW_T;  sc = rs - r * KW_T;
            } else {
                ci = k / KHWr;  const int rs = k - ci * KHWr;
                r = rs / KW;    sc = rs - r * KW;
            }
            const int ihn = ohs + r;
            const int iwn = ows + sc;
            int ih = ihn, iw = iwn;
            bool ok = mOK && (k < Kre);
            if constexpr (DIL_T == 2) {
                ok = ok && (((ihn | iwn) >= 0) && !(ihn & 1) && !(iwn & 1));
                ih = ihn >> 1; iw = iwn >> 1;
            }
            ok = ok && (ih >= 0) && (ih < H) && (iw >= 0) && (iw < W);
            const int off = ok ? ((ci * H + ih) * W + iw) : 0;  // clamped addr
            const float v = Xb[off];                            // uncond. load
            aF[e] = ok ? v : 0.f;
        }
    };
    auto loadB = [&](int k0) {
        const _Float16* src = Wp + brow + k0 + kh16;
        __builtin_prefetch(src + 2 * BK, 0, 1);        // global_prefetch_b8
        bReg = *reinterpret_cast<const v16h*>(src);
    };
    auto storeLDS = [&](int buf) {
        v8h h0, h1;
        _Float16* q0 = reinterpret_cast<_Float16*>(&h0);
        _Float16* q1 = reinterpret_cast<_Float16*>(&h1);
        #pragma unroll
        for (int e = 0; e < 8; ++e) {
            q0[e] = (_Float16)aF[e];
            q1[e] = (_Float16)aF[8 + e];
        }
        *reinterpret_cast<v8h*>(&As[buf][ra * BK + p0])   = h0;
        *reinterpret_cast<v8h*>(&As[buf][ra * BK + p1])   = h1;
        *reinterpret_cast<v16h*>(&Bs[buf][ra * BK + kh16]) = bReg;
    };

    gatherA(0); loadB(0); storeLDS(0);

    const int mlo = lane & 15;
    const int hh  = (lane >> 4) << 4;
    int cur = 0;

    for (int k0 = 0; k0 < Kp; k0 += BK) {
        __syncthreads();                 // tiles in buf[cur] are ready
        v16h af[2], bf[4];
        #pragma unroll
        for (int u = 0; u < 2; ++u)
            af[u] = *reinterpret_cast<const v16h*>(
                &As[cur][(wm * 32 + u * 16 + mlo) * BK + hh]);
        #pragma unroll
        for (int t = 0; t < 4; ++t)
            bf[t] = *reinterpret_cast<const v16h*>(
                &Bs[cur][(wn * 64 + t * 16 + mlo) * BK + hh]);

        const bool more = (k0 + BK) < Kp;
        if (more) { gatherA(k0 + BK); loadB(k0 + BK); }   // overlap with WMMA

        #pragma unroll
        for (int u = 0; u < 2; ++u)
            #pragma unroll
            for (int t = 0; t < 4; ++t)
                acc[u][t] = __builtin_amdgcn_wmma_f32_16x16x32_f16(
                    false, af[u], false, bf[t], (short)0, acc[u][t], false, false);

        if (more) { storeLDS(cur ^ 1); cur ^= 1; }  // other buffer: no barrier
    }

    // ---- epilogue (f32 C layout: VGPR v -> M = v + 8*(lane/16), N = lane%16)
    const int mbase = bm + wm * 32 + ((lane >> 4) << 3);
    const int nbase = bn + wn * 64 + (lane & 15);
    #pragma unroll
    for (int u = 0; u < 2; ++u) {
        #pragma unroll
        for (int t = 0; t < 4; ++t) {
            const int ng = nbase + t * 16;
            if (ng < N) {
                const float* ac = reinterpret_cast<const float*>(&acc[u][t]);
                #pragma unroll
                for (int vv = 0; vv < 8; ++vv) {
                    const int mg2 = mbase + u * 16 + vv;
                    if (mg2 < M) {
                        long idx;
                        if (storeMode == 0) {
                            idx = (long)mg2 * N + ng;
                        } else {
                            const int b2  = mg2 / HW;
                            const int pix = mg2 - b2 * HW;
                            idx = ((long)(b2 * N + ng)) * HW + pix;   // NCHW
                        }
                        float v = ac[vv];
                        if (bias)     v += bias[ng];
                        if (residual) v += residual[idx];
                        if (act)      v = silu_f(v);
                        Co[idx] = v;
                    }
                }
            }
        }
    }
}

// ---------------------------------------------------------------------------
// GroupNorm (32 groups) + optional SiLU. One block per (batch, group).
// ---------------------------------------------------------------------------
__global__ void __launch_bounds__(256)
group_norm_silu(const float* __restrict__ x, float* __restrict__ out,
                const float* __restrict__ scale, const float* __restrict__ bias,
                int C, int HW, int cg, int act)
{
    const int b = blockIdx.x;
    const int g = blockIdx.y;
    const long base = ((long)b * C + (long)g * cg) * HW;
    const int n = cg * HW;
    float s = 0.f, s2 = 0.f;
    for (int i = threadIdx.x; i < n; i += 256) {
        const float v = x[base + i]; s += v; s2 += v * v;
    }
    __shared__ float rs[256], rq[256];
    rs[threadIdx.x] = s; rq[threadIdx.x] = s2;
    __syncthreads();
    for (int off = 128; off > 0; off >>= 1) {
        if ((int)threadIdx.x < off) {
            rs[threadIdx.x] += rs[threadIdx.x + off];
            rq[threadIdx.x] += rq[threadIdx.x + off];
        }
        __syncthreads();
    }
    const float inv  = 1.f / (float)n;
    const float mu   = rs[0] * inv;
    const float var  = rq[0] * inv - mu * mu;
    const float rstd = rsqrtf(var + 1e-5f);
    for (int i = threadIdx.x; i < n; i += 256) {
        const int c = g * cg + i / HW;
        float v = (x[base + i] - mu) * rstd * scale[c] + bias[c];
        if (act) v = silu_f(v);
        out[base + i] = v;
    }
}

// ---------------------------------------------------------------------------
// VQ support kernels
// ---------------------------------------------------------------------------
__global__ void __launch_bounds__(256)
cb_norms(const float* __restrict__ cb, float* __restrict__ cn)
{
    const int c = blockIdx.x * blockDim.x + threadIdx.x;
    if (c < 512) {
        float s = 0.f;
        const float* r = cb + (long)c * 256;
        for (int d = 0; d < 256; ++d) { const float v = r[d]; s += v * v; }
        cn[c] = s;
    }
}

__global__ void __launch_bounds__(256)
vq_post(const float* __restrict__ z, const float* __restrict__ dots,
        const float* __restrict__ cb, const float* __restrict__ cn,
        float* __restrict__ zq, float* __restrict__ lossAcc,
        int HW, int M)
{
    const int m = blockIdx.x * blockDim.x + threadIdx.x;
    if (m >= M) return;
    const int b = m / HW, pix = m - b * HW;
    const float* zp = z + ((long)b * 256) * HW + pix;
    float zz = 0.f;
    for (int d = 0; d < 256; ++d) { const float v = zp[(long)d * HW]; zz += v * v; }
    const float* dp = dots + (long)m * 512;
    float best = 3.4e38f; int bi2 = 0;
    for (int c = 0; c < 512; ++c) {
        const float dd = zz + cn[c] - 2.f * dp[c];
        if (dd < best) { best = dd; bi2 = c; }
    }
    const float* cr = cb + (long)bi2 * 256;
    float* zqp = zq + ((long)b * 256) * HW + pix;
    float l = 0.f;
    for (int d = 0; d < 256; ++d) {
        const float cv = cr[d];
        const float df = zp[(long)d * HW] - cv;
        l += df * df;
        zqp[(long)d * HW] = cv;
    }
    atomicAdd(lossAcc, 0.25f * l / ((float)M * 256.f));
}

__global__ void zero1(float* p) { if (threadIdx.x == 0) p[0] = 0.f; }
__global__ void finalize_loss(const float* __restrict__ lossAcc, float* __restrict__ out)
{ if (threadIdx.x == 0 && blockIdx.x == 0) out[0] = lossAcc[0]; }

// ---------------------------------------------------------------------------
// Host-side helpers (all launches on the capture stream)
// ---------------------------------------------------------------------------
static inline int rup(int x, int m) { return (x + m - 1) / m * m; }

static void conv_op(hipStream_t s,
                    const float* X, int Bn, int Cin, int H, int W,
                    const float* Wsrc, int Cout, int KH, int KW,
                    int stride, int pad, int dil, int flipT,
                    _Float16* wpack,
                    float* out, int storeMode,
                    const float* residual, const float* bias, int act)
{
    const int Kre = Cin * KH * KW;
    const int Kp  = rup(Kre, BK);
    const int Nr  = rup(Cout, BN);
    {
        const int tot = Nr * Kp;
        pack_weight_kernel<<<(tot + 255) / 256, 256, 0, s>>>(
            Wsrc, wpack, Cout, Cin, KH, KW, Kp, Nr, flipT);
    }
    const int Hd = (H - 1) * dil + 1;
    const int OH = (Hd + 2 * pad - KH) / stride + 1;
    const int OW = OH;   // all tensors here are spatially square
    const int M  = Bn * OH * OW;
    dim3 grid((M + BM - 1) / BM, Nr / BN);

#define LAUNCH_GEMM(KHW_T, KW_T, DIL_T)                                        \
    wmma_conv_gemm<KHW_T, KW_T, DIL_T><<<grid, 256, 0, s>>>(                   \
        X, Bn, Cin, H, W, KH, KW, stride, pad, dil, OH, OW,                    \
        wpack, Cout, Kp, out, storeMode, residual, bias, act)

    if      (KH == 1 && KW == 1 && dil == 1) LAUNCH_GEMM(1, 1, 1);
    else if (KH == 3 && KW == 3 && dil == 1) LAUNCH_GEMM(9, 3, 1);
    else if (KH == 4 && KW == 4 && dil == 1) LAUNCH_GEMM(16, 4, 1);
    else                                     LAUNCH_GEMM(16, 4, 2);
#undef LAUNCH_GEMM
}

static void res_block(hipStream_t s, const float* Xin, int C, int Hs,
                      const float* c1, const float* c2,
                      const float* gn1s, const float* gn1b,
                      const float* gn2s, const float* gn2b,
                      _Float16* wpack, float* ta, float* tb, float* out)
{
    const int HW = Hs * Hs, cg = C / 32;
    group_norm_silu<<<dim3(64, 32), 256, 0, s>>>(Xin, ta, gn1s, gn1b, C, HW, cg, 1);
    conv_op(s, ta, 64, C, Hs, Hs, c1, C, 3, 3, 1, 1, 1, 0, wpack, tb, 1,
            nullptr, nullptr, 0);
    group_norm_silu<<<dim3(64, 32), 256, 0, s>>>(tb, ta, gn2s, gn2b, C, HW, cg, 1);
    conv_op(s, ta, 64, C, Hs, Hs, c2, C, 3, 3, 1, 1, 1, 0, wpack, out, 1,
            Xin, nullptr, 0);   // + residual
}

// ---------------------------------------------------------------------------
// kernel_launch: full forward pass.
// Input order follows jax pytree flattening (dict keys sorted):
//   params{codebooks[6], dec{coarse, head_b, head_gn_b, head_gn_s, head_w,
//          up[5]{convT, rb{conv1,conv2,gn1_b,gn1_s,gn2_b,gn2_s}, resproj}},
//          enc{down[5]{conv, rb{...}}, stem}, scale_projs[6]}, x
// ---------------------------------------------------------------------------
extern "C" void kernel_launch(void* const* d_in, const int* in_sizes, int n_in,
                              void* d_out, int out_size, void* d_ws, size_t ws_size,
                              hipStream_t stream)
{
    (void)in_sizes; (void)n_in; (void)out_size; (void)ws_size;
    auto in = [&](int i) -> const float* { return (const float*)d_in[i]; };

    const float* cbk[6];  for (int i = 0; i < 6; ++i) cbk[i] = in(i);
    const float* dec_coarse = in(6);
    const float* head_b     = in(7);
    const float* head_gn_b  = in(8);
    const float* head_gn_s  = in(9);
    const float* head_w     = in(10);
    struct UpP { const float *convT, *c1, *c2, *gn1b, *gn1s, *gn2b, *gn2s, *resproj; } up[5];
    for (int i = 0; i < 5; ++i) {
        const int b = 11 + i * 8;
        up[i] = { in(b), in(b+1), in(b+2), in(b+3), in(b+4), in(b+5), in(b+6), in(b+7) };
    }
    struct DnP { const float *conv, *c1, *c2, *gn1b, *gn1s, *gn2b, *gn2s; } dn[5];
    for (int i = 0; i < 5; ++i) {
        const int b = 51 + i * 7;
        dn[i] = { in(b), in(b+1), in(b+2), in(b+3), in(b+4), in(b+5), in(b+6) };
    }
    const float* stem = in(86);
    const float* proj[6]; for (int i = 0; i < 6; ++i) proj[i] = in(87 + i);
    const float* x = in(93);

    // ---- workspace arena ----
    char* wsp = (char*)d_ws;
    auto alloc = [&](size_t bytes) -> void* {
        void* r = (void*)wsp; wsp += (bytes + 255) & ~(size_t)255; return r;
    };
    const size_t F = sizeof(float);
    const int ich[6] = {128, 256, 512, 512, 512, 512};
    const int isp[6] = {32, 16, 8, 4, 2, 1};
    float* inter[6];
    for (int i = 0; i < 6; ++i)
        inter[i] = (float*)alloc((size_t)64 * ich[i] * isp[i] * isp[i] * F);
    const size_t ACT = (size_t)64 * 128 * 1024;   // largest activation
    float* ta = (float*)alloc(ACT * F);
    float* tb = (float*)alloc(ACT * F);
    float* t0 = (float*)alloc(ACT * F);
    float* dP = (float*)alloc(ACT * F);
    float* dQ = (float*)alloc(ACT * F);
    float* zb = (float*)alloc((size_t)64 * 256 * 1024 * F);
    float* zq[6];
    for (int i = 0; i < 6; ++i) {
        const int hw = 1 << (2 * i);
        zq[i] = (float*)alloc((size_t)64 * 256 * hw * F);
    }
    float* dotb = (float*)alloc((size_t)65536 * 512 * F);
    _Float16* wpack = (_Float16*)alloc((size_t)512 * 8192 * sizeof(_Float16));
    float* cn = (float*)alloc(512 * F);
    float* lossAcc = (float*)alloc(256);

    // ================= encoder =================
    conv_op(stream, x, 64, 3, 32, 32, stem, 128, 3, 3, 1, 1, 1, 0,
            wpack, inter[0], 1, nullptr, nullptr, 0);
    const float* h = inter[0];
    for (int i = 0; i < 5; ++i) {
        const int ci = ich[i], co = ich[i + 1], sp = isp[i];
        conv_op(stream, h, 64, ci, sp, sp, dn[i].conv, co, 4, 4, 2, 1, 1, 0,
                wpack, t0, 1, nullptr, nullptr, 0);
        res_block(stream, t0, co, isp[i + 1],
                  dn[i].c1, dn[i].c2, dn[i].gn1s, dn[i].gn1b, dn[i].gn2s, dn[i].gn2b,
                  wpack, ta, tb, inter[i + 1]);
        h = inter[i + 1];
    }

    // ================= VQ per scale =================
    zero1<<<1, 32, 0, stream>>>(lossAcc);
    for (int i = 0; i < 6; ++i) {
        const float* src = inter[5 - i];          // reversed intermediates
        const int c  = ich[5 - i];
        const int sp = isp[5 - i];                // 1,2,4,8,16,32
        conv_op(stream, src, 64, c, sp, sp, proj[5 - i], 256, 1, 1, 1, 0, 1, 0,
                wpack, zb, 1, nullptr, nullptr, 0);
        cb_norms<<<2, 256, 0, stream>>>(cbk[i], cn);
        conv_op(stream, zb, 64, 256, sp, sp, cbk[i], 512, 1, 1, 1, 0, 1, 0,
                wpack, dotb, 0, nullptr, nullptr, 0);
        const int M = 64 * sp * sp;
        vq_post<<<(M + 255) / 256, 256, 0, stream>>>(zb, dotb, cbk[i], cn,
                                                     zq[i], lossAcc, sp * sp, M);
    }

    // ================= decoder =================
    conv_op(stream, zq[0], 64, 256, 1, 1, dec_coarse, 512, 1, 1, 1, 0, 1, 0,
            wpack, dP, 1, nullptr, nullptr, 0);
    int ch = 512, sp = 1;
    const int upo[5] = {512, 512, 512, 256, 128};
    for (int i = 0; i < 5; ++i) {
        const int ic = ch, oc = upo[i];
        // conv_t(k=4, stride=2, pad=1) == conv(k=4, stride=1, pad=2, lhs_dil=2)
        conv_op(stream, dP, 64, ic, sp, sp, up[i].convT, oc, 4, 4, 1, 2, 2, 1,
                wpack, t0, 1, nullptr, nullptr, 0);
        sp *= 2; ch = oc;
        res_block(stream, t0, oc, sp,
                  up[i].c1, up[i].c2, up[i].gn1s, up[i].gn1b, up[i].gn2s, up[i].gn2b,
                  wpack, ta, tb, dQ);
        conv_op(stream, zq[i + 1], 64, 256, sp, sp, up[i].resproj, oc,
                1, 1, 1, 0, 1, 0, wpack, dP, 1, dQ, nullptr, 0);
    }

    // ================= head =================
    group_norm_silu<<<dim3(64, 32), 256, 0, stream>>>(dP, ta, head_gn_s, head_gn_b,
                                                      128, 1024, 4, 1);
    conv_op(stream, ta, 64, 128, 32, 32, head_w, 3, 3, 3, 1, 1, 1, 0,
            wpack, (float*)d_out, 1, nullptr, head_b, 0);
    finalize_loss<<<1, 32, 0, stream>>>(lossAcc, (float*)d_out + 196608);
}
